// TrajGenerator_15788299780472
// MI455X (gfx1250) — compile-verified, matrix operand order checked by reading, not run on hardware
//
#include <hip/hip_runtime.h>
#include <hip/hip_bf16.h>
#include <math.h>

#define TT 24
#define NN 1536
#define EE 64
#define DD 128   // 2*EE
#define D3 384   // 3*DD

typedef unsigned short u16;
typedef __attribute__((ext_vector_type(16))) __bf16 v16bf;
typedef __attribute__((ext_vector_type(8)))  float  v8f;

__device__ __forceinline__ u16 f2bf_bits(float f) {
    unsigned u = __float_as_uint(f);
    unsigned r = u + 0x7FFFu + ((u >> 16) & 1u);
    return (u16)(r >> 16);
}

union frag16 { uint4 q[2]; v16bf v; };

// ---------------------------------------------------------------------------
// bf16 WMMA GEMM: C[M,N] = alpha * A[M,K] x B + bias[N]
//   A row-major bf16 (lda);  B accessed as B(k,n) = B[n*ldb + k]  (operand
//   rows contiguous in k -> per-lane global_load_b128).
// ISA 16-bit fragment layout: each lane's 16 values are two contiguous
// 8-element runs: k0+half*8+[0..7] and k0+16+half*8+[0..7]  ->  2 x b128.
// K-loop is peeled: unguarded full-32 chunks + an optional K%32==16 tail
// (attention head dim) that zero-pads the high chunk. No selects -> no FLAT.
// 256 threads = 8 waves per block; each wave owns a 16x(16*NT) C tile.
// ---------------------------------------------------------------------------
template <int NT>
__global__ void wmma_gemm_kernel(const u16* __restrict__ A, int lda,
                                 const u16* __restrict__ B, int ldb,
                                 float* __restrict__ C, int ldc,
                                 int M, int N, int K,
                                 const float* __restrict__ bias, float alpha)
{
    const int wave   = threadIdx.x >> 5;
    const int lane   = threadIdx.x & 31;
    const int tile_m = blockIdx.x * 8 + wave;
    const int n0     = blockIdx.y * (16 * NT);
    const int half   = lane >> 4;
    const int l16    = lane & 15;
    const int m      = tile_m * 16 + l16;

    const uint4 z4 = {0u, 0u, 0u, 0u};
    v8f c[NT];
#pragma unroll
    for (int j = 0; j < NT; ++j) c[j] = {};

    const u16* arow = A + (size_t)m * lda + half * 8;
    const u16* brow[NT];
#pragma unroll
    for (int j = 0; j < NT; ++j)
        brow[j] = B + (size_t)(n0 + j * 16 + l16) * ldb + half * 8;

    int k0 = 0;
    for (; k0 + 32 <= K; k0 += 32) {            // full chunks: unguarded b128s
        __builtin_prefetch(arow + k0 + 64, 0, 3);   // global_prefetch_b8 (near)
        frag16 a;
        a.q[0] = *(const uint4*)(arow + k0);
        a.q[1] = *(const uint4*)(arow + k0 + 16);
#pragma unroll
        for (int j = 0; j < NT; ++j) {
            frag16 b;
            b.q[0] = *(const uint4*)(brow[j] + k0);
            b.q[1] = *(const uint4*)(brow[j] + k0 + 16);
            c[j] = __builtin_amdgcn_wmma_f32_16x16x32_bf16(false, a.v, false, b.v,
                                                           (short)0, c[j], false, false);
        }
    }
    if (k0 < K) {                                // K%32==16 tail: zero-pad high
        frag16 a;
        a.q[0] = *(const uint4*)(arow + k0);
        a.q[1] = z4;
#pragma unroll
        for (int j = 0; j < NT; ++j) {
            frag16 b;
            b.q[0] = *(const uint4*)(brow[j] + k0);
            b.q[1] = z4;
            c[j] = __builtin_amdgcn_wmma_f32_16x16x32_bf16(false, a.v, false, b.v,
                                                           (short)0, c[j], false, false);
        }
    }

#pragma unroll
    for (int j = 0; j < NT; ++j) {
        const int n  = n0 + j * 16 + l16;
        const float bn = bias ? bias[n] : 0.0f;
#pragma unroll
        for (int r = 0; r < 8; ++r) {
            const int mm = tile_m * 16 + r + half * 8;   // C layout: VGPR r -> row
            C[(size_t)mm * ldc + n] = alpha * c[j][r] + bn;
        }
    }
}

// ---------------------------------------------------------------------------
// Elementwise / reduction kernels
// ---------------------------------------------------------------------------
__global__ void embed_kernel(const float* __restrict__ x,
                             const float* __restrict__ W_loc,
                             const float* __restrict__ b_loc,
                             const float* __restrict__ time_emb,
                             float* __restrict__ p)
{
    const int tn = blockIdx.x;          // t*NN + n
    const int t  = tn / NN;
    const int c  = threadIdx.x;         // 0..127
    float v;
    if (c < EE) {
        const float x0 = x[(size_t)tn * 2 + 0];
        const float x1 = x[(size_t)tn * 2 + 1];
        v = fmaxf(x0 * W_loc[c] + x1 * W_loc[EE + c] + b_loc[c], 0.0f);
    } else {
        v = fmaxf(time_emb[t * EE + (c - EE)], 0.0f);
    }
    p[(size_t)tn * DD + c] = v;
}

__global__ void bn_stats_kernel(const float* __restrict__ p,
                                float* __restrict__ mu, float* __restrict__ istd)
{
    const int tc = blockIdx.x;          // t*DD + c
    const int t = tc / DD, c = tc % DD;
    __shared__ float s1[256], s2[256];
    float a = 0.f, b = 0.f;
    for (int n = threadIdx.x; n < NN; n += 256) {
        const float v = p[((size_t)t * NN + n) * DD + c];
        a += v; b += v * v;
    }
    s1[threadIdx.x] = a; s2[threadIdx.x] = b; __syncthreads();
    for (int s = 128; s > 0; s >>= 1) {
        if (threadIdx.x < (unsigned)s) {
            s1[threadIdx.x] += s1[threadIdx.x + s];
            s2[threadIdx.x] += s2[threadIdx.x + s];
        }
        __syncthreads();
    }
    if (threadIdx.x == 0) {
        const float m = s1[0] / NN;
        const float var = fmaxf(s2[0] / NN - m * m, 0.0f);
        mu[tc] = m; istd[tc] = rsqrtf(var + 1e-5f);
    }
}

__global__ void bn_apply_kernel(float* __restrict__ p, u16* __restrict__ pb,
                                const float* __restrict__ mu,
                                const float* __restrict__ istd,
                                const float* __restrict__ g,
                                const float* __restrict__ b)
{
    const size_t i = (size_t)blockIdx.x * blockDim.x + threadIdx.x;
    if (i >= (size_t)TT * NN * DD) return;
    const int c = (int)(i % DD);
    const int t = (int)((i / DD) / NN);
    const float v = (p[i] - mu[t * DD + c]) * istd[t * DD + c] * g[c] + b[c];
    p[i] = v; pb[i] = f2bf_bits(v);
}

__global__ void cvt_f2b_kernel(const float* __restrict__ s, u16* __restrict__ d, int n)
{
    const int i = blockIdx.x * blockDim.x + threadIdx.x;
    if (i < n) d[i] = f2bf_bits(s[i]);
}

// qkv fp32 -> bf16 (row-major) + bf16 transposed copy (for the PV operand),
// with the 1/sqrt(head_dim)=0.25 scale folded into Q.
__global__ void qkv_cvt_kernel(const float* __restrict__ qkv,
                               u16* __restrict__ qb, u16* __restrict__ qt)
{
    const int i = blockIdx.x * blockDim.x + threadIdx.x;
    if (i >= NN * D3) return;
    const int n = i / D3, c = i % D3;
    const float s = (c < DD) ? 0.25f : 1.0f;
    const u16 v = f2bf_bits(qkv[i] * s);
    qb[i] = v;
    qt[(size_t)c * NN + n] = v;
}

__global__ void softmax_row_kernel(const float* __restrict__ S, u16* __restrict__ P)
{
    const int row = blockIdx.x;
    const float* sr = S + (size_t)row * NN;
    __shared__ float red[256];
    float m = -3.0e38f;
    for (int j = threadIdx.x; j < NN; j += 256) m = fmaxf(m, sr[j]);
    red[threadIdx.x] = m; __syncthreads();
    for (int s = 128; s > 0; s >>= 1) {
        if (threadIdx.x < (unsigned)s)
            red[threadIdx.x] = fmaxf(red[threadIdx.x], red[threadIdx.x + s]);
        __syncthreads();
    }
    const float rowmax = red[0]; __syncthreads();
    float sum = 0.f;
    for (int j = threadIdx.x; j < NN; j += 256) sum += __expf(sr[j] - rowmax);
    red[threadIdx.x] = sum; __syncthreads();
    for (int s = 128; s > 0; s >>= 1) {
        if (threadIdx.x < (unsigned)s) red[threadIdx.x] += red[threadIdx.x + s];
        __syncthreads();
    }
    const float inv = 1.0f / red[0];
    for (int j = threadIdx.x; j < NN; j += 256)
        P[(size_t)row * NN + j] = f2bf_bits(__expf(sr[j] - rowmax) * inv);
}

__global__ void ln_resid_kernel(const float* __restrict__ p_t,
                                const float* __restrict__ attn,
                                const float* __restrict__ g,
                                const float* __restrict__ b,
                                float* __restrict__ st, u16* __restrict__ stb)
{
    const int n = blockIdx.x;
    const int c = threadIdx.x;              // 0..127
    __shared__ float red[128];
    const size_t o = (size_t)n * DD + c;
    const float v = p_t[o] + attn[o];
    red[c] = v; __syncthreads();
    for (int s = 64; s > 0; s >>= 1) {
        if (c < s) red[c] += red[c + s];
        __syncthreads();
    }
    const float mu = red[0] / DD; __syncthreads();
    const float d = v - mu;
    red[c] = d * d; __syncthreads();
    for (int s = 64; s > 0; s >>= 1) {
        if (c < s) red[c] += red[c + s];
        __syncthreads();
    }
    const float var = red[0] / DD;
    const float out = tanhf(d * rsqrtf(var + 1e-5f) * g[c] + b[c]);
    st[o] = out; stb[o] = f2bf_bits(out);
}

__global__ void gru_gate_kernel(const float* __restrict__ gx,
                                const float* __restrict__ gh,
                                float* __restrict__ h)
{
    const int i = blockIdx.x * blockDim.x + threadIdx.x;
    if (i >= NN * DD) return;
    const int n = i / DD, c = i % DD;
    const size_t base = (size_t)n * D3;
    const float r  = 1.f / (1.f + __expf(-(gx[base + c] + gh[base + c])));
    const float z  = 1.f / (1.f + __expf(-(gx[base + DD + c] + gh[base + DD + c])));
    const float nn = tanhf(gx[base + 2 * DD + c] + r * gh[base + 2 * DD + c]);
    h[i] = (1.f - z) * nn + z * h[i];
}

__global__ void rownorm_kernel(const float* __restrict__ src, float* __restrict__ out)
{
    const int row = blockIdx.x, c = threadIdx.x;   // 128 threads
    __shared__ float red[128];
    const float v = src[(size_t)row * DD + c];
    red[c] = v * v; __syncthreads();
    for (int s = 64; s > 0; s >>= 1) {
        if (c < s) red[c] += red[c + s];
        __syncthreads();
    }
    if (c == 0) out[row] = red[0];
}

__global__ void cdist_finish_kernel(float* __restrict__ S,
                                    const float* __restrict__ hn,
                                    const float* __restrict__ sn)
{
    const int i = blockIdx.x * blockDim.x + threadIdx.x;
    if (i >= NN * NN) return;
    const int r = i / NN, c = i % NN;
    S[i] = sqrtf(fmaxf(hn[r] + sn[c] + S[i], 0.0f));   // S already holds -2*G
}

// ---------------------------------------------------------------------------
// Deterministic auction assignment (minimize cm): single workgroup, LDS state.
// Winner = max packed (orderable-bid, row) via atomicMax => order-independent.
// ---------------------------------------------------------------------------
__device__ __forceinline__ unsigned forder(float f) {
    const unsigned u = __float_as_uint(f);
    return (u & 0x80000000u) ? ~u : (u | 0x80000000u);
}
__device__ __forceinline__ float finv(unsigned u) {
    const unsigned v = (u & 0x80000000u) ? (u & 0x7FFFFFFFu) : ~u;
    return __uint_as_float(v);
}

__global__ void auction_kernel(const float* __restrict__ cm, int* __restrict__ idx)
{
    __shared__ float price[NN];
    __shared__ int   owner[NN];
    __shared__ int   rowcol[NN];
    __shared__ unsigned long long bid[NN];
    __shared__ int   nUn;
    const int tid = threadIdx.x, nth = blockDim.x;
    for (int j = tid; j < NN; j += nth) { price[j] = 0.f; owner[j] = -1; bid[j] = 0ull; }
    for (int i = tid; i < NN; i += nth) rowcol[i] = -1;
    __syncthreads();
    const float eps = 1e-3f;
    for (int iter = 0; iter < 3000; ++iter) {
        if (tid == 0) nUn = 0;
        __syncthreads();
        for (int i = tid; i < NN; i += nth) {
            if (rowcol[i] >= 0) continue;
            atomicAdd(&nUn, 1);
            float v1 = -3.0e38f, v2 = -3.0e38f; int j1 = 0;
            const float* ci = cm + (size_t)i * NN;
            for (int j = 0; j < NN; ++j) {
                const float v = -ci[j] - price[j];
                if (v > v1) { v2 = v1; v1 = v; j1 = j; }
                else if (v > v2) v2 = v;
            }
            const float bidp = price[j1] + (v1 - v2) + eps;
            const unsigned long long pk =
                ((unsigned long long)forder(bidp) << 32) | (unsigned)i;
            atomicMax(&bid[j1], pk);
        }
        __syncthreads();
        if (nUn == 0) break;
        for (int j = tid; j < NN; j += nth) {
            const unsigned long long pk = bid[j];
            if (pk) {
                const int wi = (int)(pk & 0xFFFFFFFFull);
                const int old = owner[j];
                if (old >= 0) rowcol[old] = -1;
                owner[j] = wi; rowcol[wi] = j;
                price[j] = finv((unsigned)(pk >> 32));
                bid[j] = 0ull;
            }
        }
        __syncthreads();
    }
    if (tid == 0) {   // deterministic greedy cleanup for any stragglers
        for (int i = 0; i < NN; ++i)
            if (rowcol[i] < 0)
                for (int j = 0; j < NN; ++j)
                    if (owner[j] < 0) { owner[j] = i; rowcol[i] = j; break; }
    }
    __syncthreads();
    for (int i = tid; i < NN; i += nth) idx[i] = rowcol[i];
}

__global__ void init_kernel(const float* __restrict__ x0, const u16* __restrict__ st0b,
                            u16* __restrict__ currb, float* __restrict__ h,
                            u16* __restrict__ hb, float* __restrict__ cost,
                            float* __restrict__ gen)
{
    const int i = blockIdx.x, c = threadIdx.x;   // 128 threads
    const size_t o = (size_t)i * DD + c;
    currb[o] = st0b[o];
    h[o] = 0.f; hb[o] = 0;
    if (c == 0) cost[i] = 0.f;
    if (c < 2) gen[((size_t)i * TT) * 2 + c] = x0[(size_t)i * 2 + c];
}

__global__ void gather_step_kernel(const float* __restrict__ st,
                                   const u16* __restrict__ stb,
                                   const int* __restrict__ idx,
                                   const float* __restrict__ h,
                                   const float* __restrict__ x_t,
                                   u16* __restrict__ currb,
                                   float* __restrict__ cost,
                                   float* __restrict__ gen, int t)
{
    const int i = blockIdx.x, c = threadIdx.x;   // 128 threads
    const int j = idx[i];
    __shared__ float red[128];
    const float sv = st[(size_t)j * DD + c];
    const float d = h[(size_t)i * DD + c] - sv + 1e-6f;   // F.pairwise_distance eps
    red[c] = d * d; __syncthreads();
    for (int s = 64; s > 0; s >>= 1) {
        if (c < s) red[c] += red[c + s];
        __syncthreads();
    }
    if (c == 0) cost[i] += sqrtf(red[0]);
    currb[(size_t)i * DD + c] = stb[(size_t)j * DD + c];
    if (c < 2) gen[((size_t)i * TT + t) * 2 + c] = x_t[(size_t)j * 2 + c];
}

__global__ void final_mean_kernel(const float* __restrict__ cost, float* __restrict__ out)
{
    __shared__ float red[256];
    float s = 0.f;
    for (int i = threadIdx.x; i < NN; i += 256) s += cost[i];
    red[threadIdx.x] = s; __syncthreads();
    for (int k = 128; k > 0; k >>= 1) {
        if (threadIdx.x < (unsigned)k) red[threadIdx.x] += red[threadIdx.x + k];
        __syncthreads();
    }
    if (threadIdx.x == 0) out[NN * TT * 2] = red[0] / (float)(NN * TT);
}

// ---------------------------------------------------------------------------
extern "C" void kernel_launch(void* const* d_in, const int* in_sizes, int n_in,
                              void* d_out, int out_size, void* d_ws, size_t ws_size,
                              hipStream_t stream)
{
    (void)in_sizes; (void)n_in; (void)out_size; (void)ws_size;
    const float* x        = (const float*)d_in[0];
    const float* W_loc    = (const float*)d_in[1];
    const float* b_loc    = (const float*)d_in[2];
    const float* time_emb = (const float*)d_in[3];
    const float* bn_g     = (const float*)d_in[4];
    const float* bn_b     = (const float*)d_in[5];
    const float* in_w     = (const float*)d_in[6];
    const float* in_b     = (const float*)d_in[7];
    const float* out_w    = (const float*)d_in[8];
    const float* out_b    = (const float*)d_in[9];
    const float* ln_g     = (const float*)d_in[10];
    const float* ln_b     = (const float*)d_in[11];
    const float* w_ih     = (const float*)d_in[12];
    const float* w_hh     = (const float*)d_in[13];
    const float* b_ih     = (const float*)d_in[14];
    const float* b_hh     = (const float*)d_in[15];
    float* out = (float*)d_out;

    // workspace arena (256B aligned slots)
    char* ws = (char*)d_ws;
    size_t off = 0;
    auto alloc = [&](size_t bytes) { char* q = ws + off; off = (off + bytes + 255) & ~(size_t)255; return q; };
    const size_t TND = (size_t)TT * NN * DD;
    float* p          = (float*)alloc(TND * 4);
    u16*   p_bf       = (u16*)  alloc(TND * 2);
    float* mu         = (float*)alloc((size_t)TT * DD * 4);
    float* istd       = (float*)alloc((size_t)TT * DD * 4);
    float* states     = (float*)alloc(TND * 4);
    u16*   states_bf  = (u16*)  alloc(TND * 2);
    float* qkv        = (float*)alloc((size_t)NN * D3 * 4);
    u16*   qkv_bf     = (u16*)  alloc((size_t)NN * D3 * 2);
    u16*   qkvT_bf    = (u16*)  alloc((size_t)NN * D3 * 2);
    float* Sbuf       = (float*)alloc((size_t)NN * NN * 4);
    u16*   P_bf       = (u16*)  alloc((size_t)NN * NN * 2);
    float* o_buf      = (float*)alloc((size_t)NN * DD * 4);
    u16*   o_bf       = (u16*)  alloc((size_t)NN * DD * 2);
    float* attn_buf   = (float*)alloc((size_t)NN * DD * 4);
    u16*   in_w_bf    = (u16*)  alloc((size_t)D3 * DD * 2);
    u16*   out_w_bf   = (u16*)  alloc((size_t)DD * DD * 2);
    u16*   w_ih_bf    = (u16*)  alloc((size_t)D3 * DD * 2);
    u16*   w_hh_bf    = (u16*)  alloc((size_t)D3 * DD * 2);
    u16*   currb      = (u16*)  alloc((size_t)NN * DD * 2);
    float* h          = (float*)alloc((size_t)NN * DD * 4);
    u16*   hb         = (u16*)  alloc((size_t)NN * DD * 2);
    float* gx         = (float*)alloc((size_t)NN * D3 * 4);
    float* gh         = (float*)alloc((size_t)NN * D3 * 4);
    float* hn         = (float*)alloc((size_t)NN * 4);
    float* sn         = (float*)alloc((size_t)NN * 4);
    int*   idx        = (int*)  alloc((size_t)NN * 4);
    float* cost       = (float*)alloc((size_t)NN * 4);

    // weights -> bf16 (native [Nout,K] layout; GEMM reads B(k,n)=W[n*K+k])
    cvt_f2b_kernel<<<(D3 * DD + 255) / 256, 256, 0, stream>>>(in_w,  in_w_bf,  D3 * DD);
    cvt_f2b_kernel<<<(DD * DD + 255) / 256, 256, 0, stream>>>(out_w, out_w_bf, DD * DD);
    cvt_f2b_kernel<<<(D3 * DD + 255) / 256, 256, 0, stream>>>(w_ih,  w_ih_bf,  D3 * DD);
    cvt_f2b_kernel<<<(D3 * DD + 255) / 256, 256, 0, stream>>>(w_hh,  w_hh_bf,  D3 * DD);

    // embed + BN (batch stats over N per timestep)
    embed_kernel<<<TT * NN, 128, 0, stream>>>(x, W_loc, b_loc, time_emb, p);
    bn_stats_kernel<<<TT * DD, 256, 0, stream>>>(p, mu, istd);
    bn_apply_kernel<<<(int)((TND + 255) / 256), 256, 0, stream>>>(p, p_bf, mu, istd, bn_g, bn_b);

    const dim3 blk(256);
    // per-timestep attention
    for (int t = 0; t < TT; ++t) {
        const size_t po = (size_t)t * NN * DD;
        wmma_gemm_kernel<4><<<dim3(NN / 128, D3 / 64), blk, 0, stream>>>(
            p_bf + po, DD, in_w_bf, DD, qkv, D3, NN, D3, DD, in_b, 1.0f);
        qkv_cvt_kernel<<<(NN * D3 + 255) / 256, 256, 0, stream>>>(qkv, qkv_bf, qkvT_bf);
        for (int hd = 0; hd < 8; ++hd) {
            // S = Qs * K^T   (K dim = 16 -> tail path zero-pads to WMMA K=32)
            wmma_gemm_kernel<4><<<dim3(NN / 128, NN / 64), blk, 0, stream>>>(
                qkv_bf + hd * 16, D3, qkv_bf + DD + hd * 16, D3,
                Sbuf, NN, NN, NN, 16, nullptr, 1.0f);
            softmax_row_kernel<<<NN, 256, 0, stream>>>(Sbuf, P_bf);
            // O(:, hd*16:+16) = P * V ; V^T rows are contiguous in qkvT
            wmma_gemm_kernel<1><<<dim3(NN / 128, 1), blk, 0, stream>>>(
                P_bf, NN, qkvT_bf + (size_t)(2 * DD + hd * 16) * NN, NN,
                o_buf + hd * 16, DD, NN, 16, NN, nullptr, 1.0f);
        }
        cvt_f2b_kernel<<<(NN * DD + 255) / 256, 256, 0, stream>>>(o_buf, o_bf, NN * DD);
        wmma_gemm_kernel<4><<<dim3(NN / 128, DD / 64), blk, 0, stream>>>(
            o_bf, DD, out_w_bf, DD, attn_buf, DD, NN, DD, DD, out_b, 1.0f);
        ln_resid_kernel<<<NN, 128, 0, stream>>>(p + po, attn_buf, ln_g, ln_b,
                                                states + po, states_bf + po);
    }

    // recurrence
    init_kernel<<<NN, 128, 0, stream>>>(x, states_bf, currb, h, hb, cost, out);
    for (int t = 1; t < TT; ++t) {
        const size_t so = (size_t)t * NN * DD;
        wmma_gemm_kernel<4><<<dim3(NN / 128, D3 / 64), blk, 0, stream>>>(
            currb, DD, w_ih_bf, DD, gx, D3, NN, D3, DD, b_ih, 1.0f);
        wmma_gemm_kernel<4><<<dim3(NN / 128, D3 / 64), blk, 0, stream>>>(
            hb, DD, w_hh_bf, DD, gh, D3, NN, D3, DD, b_hh, 1.0f);
        gru_gate_kernel<<<(NN * DD + 255) / 256, 256, 0, stream>>>(gx, gh, h);
        cvt_f2b_kernel<<<(NN * DD + 255) / 256, 256, 0, stream>>>(h, hb, NN * DD);
        rownorm_kernel<<<NN, 128, 0, stream>>>(h, hn);
        rownorm_kernel<<<NN, 128, 0, stream>>>(states + so, sn);
        // Sbuf = -2 * h @ states[t]^T
        wmma_gemm_kernel<4><<<dim3(NN / 128, NN / 64), blk, 0, stream>>>(
            hb, DD, states_bf + so, DD, Sbuf, NN, NN, NN, DD, nullptr, -2.0f);
        cdist_finish_kernel<<<(NN * NN + 255) / 256, 256, 0, stream>>>(Sbuf, hn, sn);
        auction_kernel<<<1, 1024, 0, stream>>>(Sbuf, idx);
        gather_step_kernel<<<NN, 128, 0, stream>>>(states + so, states_bf + so, idx,
                                                   h, x + (size_t)t * NN * 2,
                                                   currb, cost, out, t);
    }
    final_mean_kernel<<<1, 256, 0, stream>>>(cost, out);
}
